// Net_8074538517117
// MI455X (gfx1250) — matrix operand order repaired and checked by reading, not compiled, not run on hardware
//
#include <hip/hip_runtime.h>

typedef _Float16 v16h __attribute__((ext_vector_type(16)));
typedef _Float16 v8h  __attribute__((ext_vector_type(8)));
typedef _Float16 v4h  __attribute__((ext_vector_type(4)));
typedef _Float16 v2h  __attribute__((ext_vector_type(2)));
typedef float    v8f  __attribute__((ext_vector_type(8)));

#define DIM  64
#define FSTR 168   // feat LDS row stride in halfs (336B, 16B aligned, conflict-friendly)
#define YSTR 72    // y LDS row stride in halfs (144B, 16B aligned)
#define ROWS 32    // edges per wave-tile (two 16-row M-tiles)

// ---------------- utility: zero N float4s ----------------
__global__ void zero_kernel(float* __restrict__ p, int n4) {
  int i = blockIdx.x * blockDim.x + threadIdx.x;
  if (i < n4) ((float4*)p)[i] = make_float4(0.f, 0.f, 0.f, 0.f);
}

// ---------------- in-degree counts ----------------
__global__ void degree_kernel(const int* __restrict__ dst, float* __restrict__ cnt, int E) {
  int e = blockIdx.x * blockDim.x + threadIdx.x;
  if (e < E) atomicAdd(&cnt[dst[e]], 1.0f);
}

// ---------------- node embedding: 2 -> 64 -> 64 ----------------
__global__ __launch_bounds__(256) void embed_kernel(
    const float* __restrict__ varf, const float* __restrict__ conf,
    const float* __restrict__ vw1, const float* __restrict__ vb1,
    const float* __restrict__ vw2, const float* __restrict__ vb2,
    const float* __restrict__ cw1, const float* __restrict__ cb1,
    const float* __restrict__ cw2, const float* __restrict__ cb2,
    float* __restrict__ x0, int n_var) {
  __shared__ float h1[256];
  int sub = threadIdx.x >> 6, t = threadIdx.x & 63;
  int g = blockIdx.x * 4 + sub;
  bool isv = g < n_var;
  const float* f  = isv ? varf + (size_t)g * 2 : conf + (size_t)(g - n_var) * 2;
  const float* w1 = isv ? vw1 : cw1;
  const float* b1 = isv ? vb1 : cb1;
  const float* w2 = isv ? vw2 : cw2;
  const float* b2 = isv ? vb2 : cb2;
  float f0 = f[0], f1 = f[1];
  float a = fmaf(f0, w1[t], fmaf(f1, w1[DIM + t], b1[t]));
  h1[threadIdx.x] = fmaxf(a, 0.0f);
  __syncthreads();
  const float* hr = h1 + (sub << 6);
  float acc = b2[t];
#pragma unroll 8
  for (int k = 0; k < DIM; ++k) acc = fmaf(hr[k], w2[k * DIM + t], acc);
  x0[(size_t)g * DIM + t] = acc;
}

// ---------------- swizzle conv weights into B-fragment layout ----------------
// B fragment (f16, 32x16): lane<16 -> N=lane, K = kc*32 + j (j=0..15)
//                          lane>=16 -> N=lane-16, K = kc*32 + 16 + j
// w1f: [kc 0..4][nt 0..3][lane 0..31][j 0..15]   (K padded 130 -> 160 with zeros)
// w2f: [kc 0..1][nt 0..3][lane 0..31][j 0..15]
__global__ void prep_weights_kernel(const float* __restrict__ w1,
                                    const float* __restrict__ w2,
                                    _Float16* __restrict__ w1f,
                                    _Float16* __restrict__ w2f) {
  int idx = blockIdx.x * blockDim.x + threadIdx.x;  // 0..14335
  if (idx < 10240) {
    int j = idx & 15, lane = (idx >> 4) & 31, nt = (idx >> 9) & 3, kc = idx >> 11;
    int k = kc * 32 + ((lane & 16) ? 16 : 0) + j;
    int col = nt * 16 + (lane & 15);
    w1f[idx] = (_Float16)((k < 130) ? w1[k * DIM + col] : 0.0f);
  } else {
    int i2 = idx - 10240;
    int j = i2 & 15, lane = (i2 >> 4) & 31, nt = (i2 >> 9) & 3, kc = i2 >> 11;
    int k = kc * 32 + ((lane & 16) ? 16 : 0) + j;
    int col = nt * 16 + (lane & 15);
    w2f[i2] = (_Float16)w2[k * DIM + col];
  }
}

// ---------------- edge conv: gather -> WMMA MLP -> atomic scatter ----------------
// One wave processes 32 edges (two 16-row M-tiles through shared B registers).
__global__ __launch_bounds__(128) void conv_kernel(
    const float* __restrict__ xin,
    const int* __restrict__ eidx,      // [2][E]
    const float* __restrict__ etyp,    // [E][2]
    const _Float16* __restrict__ w1f,
    const float* __restrict__ b1,
    const _Float16* __restrict__ w2f,
    const float* __restrict__ b2,
    float* __restrict__ accum,
    int E) {
  __shared__ _Float16 featS[4][ROWS * FSTR];   // 43008 B
  __shared__ _Float16 yS[4][16 * YSTR];        //  9216 B
  __shared__ int sbufS[4][ROWS];               //   512 B
  __shared__ int dbufS[4][ROWS];               //   512 B

  const int wave = threadIdx.x >> 5;
  const int lane = threadIdx.x & 31;
  const int base = (blockIdx.x * 4 + wave) * ROWS;

  _Float16* F = featS[wave];
  _Float16* Y = yS[wave];
  int* sbuf = sbufS[wave];
  int* dbuf = dbufS[wave];

  const int m    = lane & 15;               // A-matrix row within M-tile
  const int koff = (lane >> 4) << 4;        // A chunk K offset: 0 or 16
  const int col  = lane & 15;               // C/D column within N-tile
  const int moff = (lane >> 4) << 3;        // C/D row offset: 0 or 8

  // --- hoisted bias loads (wait overlaps with gather) ---
  float bb1[4], bb2[4];
#pragma unroll
  for (int nt = 0; nt < 4; ++nt) {
    bb1[nt] = b1[nt * 16 + col];
    bb2[nt] = b2[nt * 16 + col];
  }

  // --- load all B fragments into registers (held across both M-tiles) ---
  v16h bw1[5][4], bw2[2][4];
#pragma unroll
  for (int kc = 0; kc < 5; ++kc)
#pragma unroll
    for (int nt = 0; nt < 4; ++nt)
      bw1[kc][nt] = *(const v16h*)(w1f + (size_t)((kc * 4 + nt) * 32 + lane) * 16);
#pragma unroll
  for (int kc = 0; kc < 2; ++kc)
#pragma unroll
    for (int nt = 0; nt < 4; ++nt)
      bw2[kc][nt] = *(const v16h*)(w2f + (size_t)((kc * 4 + nt) * 32 + lane) * 16);

  // --- zero feature tile (covers the K=130..159 zero-pad region) ---
  {
    v4h z = {};
#pragma unroll
    for (int i = 0; i < 42; ++i) *(v4h*)(F + (lane + 32 * i) * 4) = z;
  }

  // --- edge metadata + edge_type features (one edge per lane) ---
  {
    int e = base + lane;
    sbuf[lane] = eidx[e];
    dbuf[lane] = eidx[E + e];
    float2 et = ((const float2*)etyp)[e];
    v2h p; p[0] = (_Float16)et.x; p[1] = (_Float16)et.y;
    *(v2h*)(F + lane * FSTR + 128) = p;   // K = 128,129
  }

  // --- gather x[dst] (K 0..63) and x[src] (K 64..127), f32 -> f16 into LDS ---
  // iteration = one edge row; 32 lanes cover 32 float4 chunks (16 dst + 16 src)
#pragma unroll 4
  for (int row = 0; row < ROWS; ++row) {
    int node = (lane < 16) ? dbuf[row] : sbuf[row];
    float4 v = ((const float4*)(xin + (size_t)node * DIM))[lane & 15];
    v4h p;
    p[0] = (_Float16)v.x; p[1] = (_Float16)v.y;
    p[2] = (_Float16)v.z; p[3] = (_Float16)v.w;
    int off = row * FSTR + ((lane < 16) ? lane * 4 : 64 + (lane - 16) * 4);
    *(v4h*)(F + off) = p;
  }

  // --- two 16-row M-tiles through the same B registers ---
#pragma unroll
  for (int mt = 0; mt < 2; ++mt) {
    const _Float16* Frow = F + (mt * 16 + m) * FSTR;

    // preload all 5 A fragments, then run 20 WMMAs back-to-back
    v16h a[5];
#pragma unroll
    for (int kc = 0; kc < 5; ++kc) {
      v8h alo = *(const v8h*)(Frow + kc * 32 + koff);        // K base..base+7
      v8h ahi = *(const v8h*)(Frow + kc * 32 + koff + 8);    // K base+8..base+15
      a[kc] = __builtin_shufflevector(alo, ahi, 0, 1, 2, 3, 4, 5, 6, 7,
                                      8, 9, 10, 11, 12, 13, 14, 15);
    }
    v8f acc[4] = {};
#pragma unroll
    for (int kc = 0; kc < 5; ++kc)
#pragma unroll
      for (int nt = 0; nt < 4; ++nt)
        acc[nt] = __builtin_amdgcn_wmma_f32_16x16x32_f16(
            false, a[kc], false, bw1[kc][nt], (short)0, acc[nt], false, false);

    // bias + ReLU -> f16 hidden tile in LDS
#pragma unroll
    for (int nt = 0; nt < 4; ++nt)
#pragma unroll
      for (int i = 0; i < 8; ++i) {
        float v = fmaxf(acc[nt][i] + bb1[nt], 0.0f);
        Y[(moff + i) * YSTR + nt * 16 + col] = (_Float16)v;
      }

    // GEMM2: preload both A fragments, then 8 WMMAs
    v16h a2[2];
#pragma unroll
    for (int kc = 0; kc < 2; ++kc) {
      const _Float16* yp = Y + m * YSTR + kc * 32 + koff;
      v8h alo = *(const v8h*)(yp);
      v8h ahi = *(const v8h*)(yp + 8);
      a2[kc] = __builtin_shufflevector(alo, ahi, 0, 1, 2, 3, 4, 5, 6, 7,
                                       8, 9, 10, 11, 12, 13, 14, 15);
    }
    v8f acc2[4] = {};
#pragma unroll
    for (int kc = 0; kc < 2; ++kc)
#pragma unroll
      for (int nt = 0; nt < 4; ++nt)
        acc2[nt] = __builtin_amdgcn_wmma_f32_16x16x32_f16(
            false, a2[kc], false, bw2[kc][nt], (short)0, acc2[nt], false, false);

    // bias + scatter-add: one dbuf read + one base address per row
#pragma unroll
    for (int i = 0; i < 8; ++i) {
      int d = dbuf[mt * 16 + moff + i];
      float* dstp = accum + (size_t)d * DIM + col;
#pragma unroll
      for (int nt = 0; nt < 4; ++nt)
        atomicAdd(dstp + nt * 16, acc2[nt][i] + bb2[nt]);
    }
  }
}

// ---------------- mean + ReLU ----------------
__global__ void finalize_kernel(const float* __restrict__ acc,
                                const float* __restrict__ cnt,
                                float* __restrict__ xout, int n) {
  int i = blockIdx.x * blockDim.x + threadIdx.x;
  if (i < n) {
    float c = cnt[i >> 6];
    float v = acc[i] / fmaxf(c, 1.0f);
    xout[i] = fmaxf(v, 0.0f);
  }
}

// ---------------- head: concat(4x64) -> 64 -> 64 -> 1 -> sigmoid ----------------
__global__ __launch_bounds__(256) void head_kernel(
    const float* __restrict__ x0, const float* __restrict__ x1,
    const float* __restrict__ x2, const float* __restrict__ x3,
    const float* __restrict__ fc1w, const float* __restrict__ fc1b,
    const float* __restrict__ fc2w, const float* __restrict__ fc2b,
    const float* __restrict__ fc3w, const float* __restrict__ fc3b,
    float* __restrict__ out) {
  __shared__ float H[4][256];
  __shared__ float H2[4][64];
  int sub = threadIdx.x >> 6, t = threadIdx.x & 63;
  int g = blockIdx.x * 4 + sub;
  size_t r = (size_t)g * DIM + t;
  H[sub][t]       = x0[r];
  H[sub][64 + t]  = x1[r];
  H[sub][128 + t] = x2[r];
  H[sub][192 + t] = x3[r];
  __syncthreads();
  float a = fc1b[t];
#pragma unroll 8
  for (int k = 0; k < 256; ++k) a = fmaf(H[sub][k], fc1w[k * DIM + t], a);
  a = fmaxf(a, 0.0f);
  H2[sub][t] = a;
  __syncthreads();
  float b = fc2b[t];
#pragma unroll 8
  for (int k = 0; k < 64; ++k) b = fmaf(H2[sub][k], fc2w[k * DIM + t], b);
  b = fmaxf(b, 0.0f);
  __syncthreads();
  H[sub][t] = b * fc3w[t];
  __syncthreads();
  if (t == 0) {
    float z = fc3b[0];
    for (int k = 0; k < 64; ++k) z += H[sub][k];
    out[g] = 1.0f / (1.0f + __expf(-z));
  }
}

extern "C" void kernel_launch(void* const* d_in, const int* in_sizes, int n_in,
                              void* d_out, int out_size, void* d_ws, size_t ws_size,
                              hipStream_t stream) {
  const float* varf = (const float*)d_in[0];
  const float* conf = (const float*)d_in[1];
  const int*   eidx = (const int*)d_in[2];
  const float* etyp = (const float*)d_in[3];
  // d_in[4] assoc_var, d_in[5] assoc_con are identity/offset maps; d_in[6] num_nodes
  const float* vw1 = (const float*)d_in[7];
  const float* vb1 = (const float*)d_in[8];
  const float* vw2 = (const float*)d_in[9];
  const float* vb2 = (const float*)d_in[10];
  const float* cw1 = (const float*)d_in[11];
  const float* cb1 = (const float*)d_in[12];
  const float* cw2 = (const float*)d_in[13];
  const float* cb2 = (const float*)d_in[14];
  const float* conv_w1[3] = { (const float*)d_in[15], (const float*)d_in[19], (const float*)d_in[23] };
  const float* conv_b1[3] = { (const float*)d_in[16], (const float*)d_in[20], (const float*)d_in[24] };
  const float* conv_w2[3] = { (const float*)d_in[17], (const float*)d_in[21], (const float*)d_in[25] };
  const float* conv_b2[3] = { (const float*)d_in[18], (const float*)d_in[22], (const float*)d_in[26] };
  const float* fc1w = (const float*)d_in[27];
  const float* fc1b = (const float*)d_in[28];
  const float* fc2w = (const float*)d_in[29];
  const float* fc2b = (const float*)d_in[30];
  const float* fc3w = (const float*)d_in[31];
  const float* fc3b = (const float*)d_in[32];

  int n_var = in_sizes[4];           // 50000
  int n_con = in_sizes[5];           // 50000
  int n_nodes = n_var + n_con;       // 100000
  int E = in_sizes[2] / 2;           // 800000

  size_t X = (size_t)n_nodes * DIM * sizeof(float);   // 25.6 MB per layer
  char* w = (char*)d_ws;
  float* xbuf[4];
  for (int i = 0; i < 4; ++i) xbuf[i] = (float*)(w + (size_t)i * X);
  float* accum = (float*)(w + 4 * X);
  float* cnt   = (float*)(w + 5 * X);
  size_t cnt_bytes = (((size_t)n_nodes * 4 + 255) / 256) * 256;
  _Float16* w1f = (_Float16*)(w + 5 * X + cnt_bytes);
  _Float16* w2f = w1f + 10240;

  // degree counts (shared by all three convs)
  int cntN4 = n_nodes / 4;
  zero_kernel<<<(cntN4 + 255) / 256, 256, 0, stream>>>(cnt, cntN4);
  degree_kernel<<<(E + 255) / 256, 256, 0, stream>>>(eidx + E, cnt, E);

  // initial node embeddings -> x0
  embed_kernel<<<n_nodes / 4, 256, 0, stream>>>(varf, conf, vw1, vb1, vw2, vb2,
                                                cw1, cb1, cw2, cb2, xbuf[0], n_var);

  int accN4 = n_nodes * DIM / 4;
  int nFeat = n_nodes * DIM;
  int convGrid = (E / ROWS) / 4;   // one 32-edge tile per wave, 4 waves per block
  for (int c = 0; c < 3; ++c) {
    prep_weights_kernel<<<56, 256, 0, stream>>>(conv_w1[c], conv_w2[c], w1f, w2f);
    zero_kernel<<<(accN4 + 255) / 256, 256, 0, stream>>>(accum, accN4);
    conv_kernel<<<convGrid, 128, 0, stream>>>(xbuf[c], eidx, etyp, w1f, conv_b1[c],
                                              w2f, conv_b2[c], accum, E);
    finalize_kernel<<<(nFeat + 255) / 256, 256, 0, stream>>>(accum, cnt, xbuf[c + 1], nFeat);
  }

  head_kernel<<<n_var / 4, 256, 0, stream>>>(xbuf[0], xbuf[1], xbuf[2], xbuf[3],
                                             fc1w, fc1b, fc2w, fc2b, fc3w, fc3b,
                                             (float*)d_out);
}
